// SFM_v5_75136157876812
// MI455X (gfx1250) — matrix operand order repaired and checked by reading, not compiled, not run on hardware
//
#include <hip/hip_runtime.h>
#include <hip/hip_bf16.h>
#include <cstdint>

// ---------------------------------------------------------------------------
// MI455X (gfx1250). Compute-bound GEMM workload (~287 GFLOP vs ~120MB traffic
// => far above the 23.3TB/s roofline knee). All GEMMs run on
// v_wmma_f32_16x16x32_bf16 (bf16 inputs, f32 accumulate). All f32->bf16
// rounding is done OUTSIDE the GEMM (one-time weight/activation converts,
// producers emit bf16 directly), so the GEMM inner loop is pure
// global_load_b128 / ds_store_b128 / ds_load_b128 / v_wmma with LDS
// double-buffering and one barrier per K-slice.
// ---------------------------------------------------------------------------

typedef __attribute__((ext_vector_type(16))) __bf16        bf16x16;
typedef __attribute__((ext_vector_type(8)))  float         floatx8;
typedef __attribute__((ext_vector_type(4)))  unsigned int  uintx4;

struct Frag32B { uintx4 lo, hi; };   // 32 bytes == one v16bf fragment

__device__ __forceinline__ unsigned bf16_rne(float f) {
  unsigned u = __float_as_uint(f);
  return (u + 0x7FFFu + ((u >> 16) & 1u)) >> 16;   // round-to-nearest-even
}
__device__ __forceinline__ unsigned pack_bf16x2(float lo, float hi) {
  return bf16_rne(lo) | (bf16_rne(hi) << 16);
}

#define BM 128
#define BN 64
#define BK 32
#define LDS_ROW_U 20   // 16 data uints (32 bf16) + 4 pad; 80B rows, 16B aligned

// C[M,N] = A[M,K]*W[N,K]^T (+bias[N]) (+C).  A,W are bf16; M%128==0, N%64==0,
// K%32==0.  256 threads = 8 waves (4x2), each wave owns a 32x32 patch.
__global__ __launch_bounds__(256)
void gemm_bf16_wmma(const unsigned short* __restrict__ A,
                    const unsigned short* __restrict__ W,
                    const float* __restrict__ bias, float* C,
                    int N, int K, int has_bias, int accumulate) {
  __shared__ __align__(16) unsigned As[2][BM * LDS_ROW_U];
  __shared__ __align__(16) unsigned Bs[2][BN * LDS_ROW_U];

  const int tid  = threadIdx.x;
  const int lane = tid & 31;
  const int wave = tid >> 5;      // 0..7
  const int wm   = wave >> 1;     // 0..3  (M dir, 32 rows each)
  const int wn   = wave & 1;      // 0..1  (N dir, 32 cols each)
  const int half = lane >> 4;
  const int l15  = lane & 15;

  const int m0 = blockIdx.y * BM;
  const int n0 = blockIdx.x * BN;

  // tile-staging assignment: 4 threads/row, 16B (8 bf16) each
  const int srow = tid >> 2;         // 0..63
  const int scu  = (tid & 3) * 4;    // uint offset 0,4,8,12 within 16-uint row

  const size_t arow0 = (size_t)(m0 + srow) * K;        // A pass 0 row
  const size_t arow1 = (size_t)(m0 + srow + 64) * K;   // A pass 1 row
  const size_t brow  = (size_t)(n0 + srow) * K;        // B row
  const int    ecol  = scu * 2;                        // bf16 element col

  floatx8 acc00 = 0.f, acc01 = 0.f, acc10 = 0.f, acc11 = 0.f;

  // ---- prologue: fetch K-slice 0 and stage into buffer 0 ----
  uintx4 ra0 = *(const uintx4*)(A + arow0 + ecol);
  uintx4 ra1 = *(const uintx4*)(A + arow1 + ecol);
  uintx4 rb  = *(const uintx4*)(W + brow  + ecol);
  *(uintx4*)(&As[0][srow * LDS_ROW_U + scu])        = ra0;
  *(uintx4*)(&As[0][(srow + 64) * LDS_ROW_U + scu]) = ra1;
  *(uintx4*)(&Bs[0][srow * LDS_ROW_U + scu])        = rb;
  __syncthreads();

  const int abase = half ? 4 : 0;   // A frag: lanes16-31 -> K{8..15,24..31}
  const int bbase = half ? 4 : 0;   // B frag: lanes16-31 -> K{8..15,24..31}

  int buf = 0;
  for (int k0 = 0; k0 < K; k0 += BK, buf ^= 1) {
    const bool more = (k0 + BK) < K;
    // issue next slice's global loads early (hide latency under WMMA)
    if (more) {
      const int kn = k0 + BK;
      ra0 = *(const uintx4*)(A + arow0 + kn + ecol);
      ra1 = *(const uintx4*)(A + arow1 + kn + ecol);
      rb  = *(const uintx4*)(W + brow  + kn + ecol);
      if (k0 + 2 * BK < K) {   // global_prefetch_b8 for slice k+2
        __builtin_prefetch(A + arow0 + k0 + 2 * BK + ecol, 0, 1);
        __builtin_prefetch(W + brow  + k0 + 2 * BK + ecol, 0, 1);
      }
    }

    // ---- fragments from current buffer (documented CDNA5 layouts) ----
    Frag32B fa0, fa1, fb0, fb1;
    {
      const unsigned* pr = &As[buf][(wm * 32 + l15) * LDS_ROW_U];
      fa0.lo = *(const uintx4*)(pr + abase);
      fa0.hi = *(const uintx4*)(pr + abase + 8);
      pr = &As[buf][(wm * 32 + 16 + l15) * LDS_ROW_U];
      fa1.lo = *(const uintx4*)(pr + abase);
      fa1.hi = *(const uintx4*)(pr + abase + 8);
      const unsigned* pc = &Bs[buf][(wn * 32 + l15) * LDS_ROW_U];
      fb0.lo = *(const uintx4*)(pc + bbase);
      fb0.hi = *(const uintx4*)(pc + bbase + 8);
      pc = &Bs[buf][(wn * 32 + 16 + l15) * LDS_ROW_U];
      fb1.lo = *(const uintx4*)(pc + bbase);
      fb1.hi = *(const uintx4*)(pc + bbase + 8);
    }
    const bf16x16 a0 = __builtin_bit_cast(bf16x16, fa0);
    const bf16x16 a1 = __builtin_bit_cast(bf16x16, fa1);
    const bf16x16 b0 = __builtin_bit_cast(bf16x16, fb0);
    const bf16x16 b1 = __builtin_bit_cast(bf16x16, fb1);

    acc00 = __builtin_amdgcn_wmma_f32_16x16x32_bf16(false, a0, false, b0, (short)0, acc00, false, false);
    acc01 = __builtin_amdgcn_wmma_f32_16x16x32_bf16(false, a0, false, b1, (short)0, acc01, false, false);
    acc10 = __builtin_amdgcn_wmma_f32_16x16x32_bf16(false, a1, false, b0, (short)0, acc10, false, false);
    acc11 = __builtin_amdgcn_wmma_f32_16x16x32_bf16(false, a1, false, b1, (short)0, acc11, false, false);

    // stage next slice into the other buffer; single barrier per K-step
    if (more) {
      const int nb = buf ^ 1;
      *(uintx4*)(&As[nb][srow * LDS_ROW_U + scu])        = ra0;
      *(uintx4*)(&As[nb][(srow + 64) * LDS_ROW_U + scu]) = ra1;
      *(uintx4*)(&Bs[nb][srow * LDS_ROW_U + scu])        = rb;
    }
    __syncthreads();
  }

  // ---- epilogue: C/D layout => VGPR r, lanes0-15: M=r, lanes16-31: M=r+8 ----
  const int gm_base = m0 + wm * 32;
  const int gn0 = n0 + wn * 32 + l15;
  const int gn1 = gn0 + 16;
  float bv0 = 0.f, bv1 = 0.f;
  if (has_bias) { bv0 = bias[gn0]; bv1 = bias[gn1]; }

  #pragma unroll
  for (int r = 0; r < 8; ++r) {
    const int gmA = gm_base + half * 8 + r;
    const int gmB = gmA + 16;
    size_t o;
    float v;
    o = (size_t)gmA * N + gn0; v = acc00[r] + bv0; if (accumulate) v += C[o]; C[o] = v;
    o = (size_t)gmA * N + gn1; v = acc01[r] + bv1; if (accumulate) v += C[o]; C[o] = v;
    o = (size_t)gmB * N + gn0; v = acc10[r] + bv0; if (accumulate) v += C[o]; C[o] = v;
    o = (size_t)gmB * N + gn1; v = acc11[r] + bv1; if (accumulate) v += C[o]; C[o] = v;
  }
}

// ---------------------------------------------------------------------------
__global__ __launch_bounds__(256)
void cvt_f32_to_bf16(const float* __restrict__ src, unsigned short* __restrict__ dst,
                     size_t n4) {   // n4 = count/4
  const size_t i = (size_t)blockIdx.x * 256 + threadIdx.x;
  if (i >= n4) return;
  const float4 v = ((const float4*)src)[i];
  uint2 o;
  o.x = pack_bf16x2(v.x, v.y);
  o.y = pack_bf16x2(v.z, v.w);
  ((uint2*)dst)[i] = o;
}

// out_bf16[row] = bf16( base[row] + a[row] * ((b1.a)+(b2.a))/512 )
__global__ __launch_bounds__(256)
void bilin_update_bf16(const float* __restrict__ base, const float* __restrict__ a,
                       const float* __restrict__ b1, const float* __restrict__ b2,
                       unsigned short* __restrict__ outbf) {
  const int row = blockIdx.x;
  const size_t off = (size_t)row * 512;
  const int i0 = threadIdx.x * 2;        // 256 threads x 2 elems = 512
  const float a0 = a[off + i0], a1 = a[off + i0 + 1];
  float s = a0 * b1[off + i0] + a1 * b1[off + i0 + 1]
          + a0 * b2[off + i0] + a1 * b2[off + i0 + 1];
  #pragma unroll
  for (int m = 16; m > 0; m >>= 1) s += __shfl_xor(s, m, 32);
  __shared__ float sh[8];
  if ((threadIdx.x & 31) == 0) sh[threadIdx.x >> 5] = s;
  __syncthreads();
  float tot = 0.f;
  #pragma unroll
  for (int i = 0; i < 8; ++i) tot += sh[i];
  const float scale = tot * (1.0f / 512.0f);
  const float v0 = base[off + i0]     + a0 * scale;
  const float v1 = base[off + i0 + 1] + a1 * scale;
  ((unsigned*)outbf)[(size_t)row * 256 + threadIdx.x] = pack_bf16x2(v0, v1);
}

__global__ __launch_bounds__(256)
void zero_u32(unsigned* p, size_t n) {
  const size_t i = (size_t)blockIdx.x * 256 + threadIdx.x;
  if (i < n) p[i] = 0u;
}

// g[B,2048] gates (i,f,g,o); update c (f32), h (bf16 for next GEMM), fout (f32)
__global__ __launch_bounds__(256)
void lstm_gates(const float* __restrict__ g, unsigned short* __restrict__ hbf,
                float* __restrict__ c, float* __restrict__ fout) {
  const int idx = blockIdx.x * 256 + threadIdx.x;   // b*512 + j
  const int b = idx >> 9;
  const int j = idx & 511;
  const float* gr = g + (size_t)b * 2048;
  const float i_ = gr[j];
  const float f_ = gr[512 + j];
  const float g_ = gr[1024 + j];
  const float o_ = gr[1536 + j];
  const float si = 1.f / (1.f + __expf(-i_));
  const float sf = 1.f / (1.f + __expf(-f_));
  const float so = 1.f / (1.f + __expf(-o_));
  float cc = c[idx];
  cc = sf * cc + si * tanhf(g_);
  const float hh = so * tanhf(cc);
  c[idx] = cc;
  hbf[idx] = (unsigned short)bf16_rne(hh);
  fout[idx] = hh;
}

// Fused heads + 4-way softmax: per (b,oc) 3x512 + 1x1536 dot products.
__global__ __launch_bounds__(256)
void head_softmax(const float* __restrict__ fx, const float* __restrict__ fy,
                  const float* __restrict__ fz,
                  const float* __restrict__ wcx, const float* __restrict__ bcx,
                  const float* __restrict__ wcy, const float* __restrict__ bcy,
                  const float* __restrict__ wcz, const float* __restrict__ bcz,
                  const float* __restrict__ wcxyz, const float* __restrict__ bcxyz,
                  float* __restrict__ out, int Bsz) {
  const int idx = blockIdx.x * 256 + threadIdx.x;   // b*26 + o
  if (idx >= Bsz * 26) return;
  const int b = idx / 26;
  const int o = idx - b * 26;

  const float* fxr = fx + (size_t)b * 512;
  const float* fyr = fy + (size_t)b * 512;
  const float* fzr = fz + (size_t)b * 512;
  const float* w0 = wcx + (size_t)o * 512;
  const float* w1 = wcy + (size_t)o * 512;
  const float* w2 = wcz + (size_t)o * 512;
  const float* w3 = wcxyz + (size_t)o * 1536;

  float v0 = bcx[o], v1 = bcy[o], v2 = bcz[o], v3 = bcxyz[o];
  for (int k = 0; k < 512; ++k) {
    const float ax = fxr[k], ay = fyr[k], az = fzr[k];
    v0 = fmaf(ax, w0[k], v0);
    v1 = fmaf(ay, w1[k], v1);
    v2 = fmaf(az, w2[k], v2);
    v3 = fmaf(ax, w3[k], fmaf(ay, w3[512 + k], fmaf(az, w3[1024 + k], v3)));
  }
  const float m = fmaxf(fmaxf(v0, v1), fmaxf(v2, v3));
  const float e0 = __expf(v0 - m), e1 = __expf(v1 - m);
  const float e2 = __expf(v2 - m), e3 = __expf(v3 - m);
  const float inv = 1.f / (e0 + e1 + e2 + e3);
  const size_t plane = (size_t)Bsz * 26;
  out[idx]             = e0 * inv;
  out[plane + idx]     = e1 * inv;
  out[2 * plane + idx] = e2 * inv;
  out[3 * plane + idx] = e3 * inv;
}

// ---------------------------------------------------------------------------
extern "C" void kernel_launch(void* const* d_in, const int* in_sizes, int n_in,
                              void* d_out, int out_size, void* d_ws, size_t ws_size,
                              hipStream_t stream) {
  constexpr int Bsz = 16384, IC = 512;
  const float* x    = (const float*)d_in[0];
  const float* y    = (const float*)d_in[1];
  const float* z    = (const float*)d_in[2];
  const float* wx1  = (const float*)d_in[3];  const float* bx1 = (const float*)d_in[4];
  const float* wx2  = (const float*)d_in[5];  const float* bx2 = (const float*)d_in[6];
  const float* wx3  = (const float*)d_in[7];  const float* bx3 = (const float*)d_in[8];
  const float* wy1  = (const float*)d_in[9];  const float* by1 = (const float*)d_in[10];
  const float* wy2  = (const float*)d_in[11]; const float* by2 = (const float*)d_in[12];
  const float* wy3  = (const float*)d_in[13]; const float* by3 = (const float*)d_in[14];
  const float* wz1  = (const float*)d_in[15]; const float* bz1 = (const float*)d_in[16];
  const float* wz2  = (const float*)d_in[17]; const float* bz2 = (const float*)d_in[18];
  const float* wz3  = (const float*)d_in[19]; const float* bz3 = (const float*)d_in[20];
  const float* wcx  = (const float*)d_in[21]; const float* bcx = (const float*)d_in[22];
  const float* wcy  = (const float*)d_in[23]; const float* bcy = (const float*)d_in[24];
  const float* wcz  = (const float*)d_in[25]; const float* bcz = (const float*)d_in[26];
  const float* wcxyz= (const float*)d_in[27]; const float* bcxyz=(const float*)d_in[28];
  const float* w_ih = (const float*)d_in[29];
  const float* w_hh = (const float*)d_in[30];
  float* out = (float*)d_out;

  float* ws = (float*)d_ws;
  const size_t U = (size_t)Bsz * IC;        // 8.39M elements
  const size_t WSQ = (size_t)IC * IC;       // 262144
  const size_t WL  = (size_t)(4 * IC) * IC; // 1048576

  float* t1 = ws;          float* t2 = ws + U;  float* t3 = ws + 2 * U;
  float* c  = ws + 3 * U;
  float* g  = ws + 4 * U;                    // [B,2048] = 4U
  unsigned short* bfb  = (unsigned short*)(ws + 8 * U);
  unsigned short* xbf  = bfb;            unsigned short* ybf  = xbf  + U;
  unsigned short* zbf  = ybf  + U;       unsigned short* xubf = zbf  + U;
  unsigned short* yubf = xubf + U;       unsigned short* zubf = yubf + U;
  unsigned short* hbf  = zubf + U;
  unsigned short* wq[9];
  wq[0] = hbf + U;
  for (int i = 1; i < 9; ++i) wq[i] = wq[i - 1] + WSQ;
  unsigned short* wihbf = wq[8] + WSQ;
  unsigned short* whhbf = wihbf + WL;

  const dim3 blk(256);
  auto cvt = [&](const float* s, unsigned short* d, size_t n) {
    const size_t n4 = n / 4;
    hipLaunchKernelGGL(cvt_f32_to_bf16, dim3((unsigned)((n4 + 255) / 256)), blk, 0,
                       stream, s, d, n4);
  };
  auto gemm = [&](const unsigned short* Ap, const unsigned short* Wp,
                  const float* bp, float* Cp, int N, int hb, int acm) {
    dim3 grid(N / BN, Bsz / BM);
    hipLaunchKernelGGL(gemm_bf16_wmma, grid, blk, 0, stream,
                       Ap, Wp, bp, Cp, N, IC, hb, acm);
  };

  // ---- one-time (per launch) bf16 conversions ----
  cvt(x, xbf, U); cvt(y, ybf, U); cvt(z, zbf, U);
  const float* wsq[9] = {wx1, wx2, wx3, wy1, wy2, wy3, wz1, wz2, wz3};
  for (int i = 0; i < 9; ++i) cvt(wsq[i], wq[i], WSQ);
  cvt(w_ih, wihbf, WL);
  cvt(w_hh, whhbf, WL);

  // -------- bilinear mixing stages (x, then y w/ updated x, then z) --------
  gemm(xbf, wq[0], bx1, t1, IC, 1, 0);
  gemm(ybf, wq[1], bx2, t2, IC, 1, 0);
  gemm(zbf, wq[2], bx3, t3, IC, 1, 0);
  hipLaunchKernelGGL(bilin_update_bf16, dim3(Bsz), blk, 0, stream, x, t1, t2, t3, xubf);

  gemm(ybf,  wq[3], by1, t1, IC, 1, 0);
  gemm(zbf,  wq[4], by2, t2, IC, 1, 0);
  gemm(xubf, wq[5], by3, t3, IC, 1, 0);
  hipLaunchKernelGGL(bilin_update_bf16, dim3(Bsz), blk, 0, stream, y, t1, t2, t3, yubf);

  gemm(zbf,  wq[6], bz1, t1, IC, 1, 0);
  gemm(yubf, wq[7], bz2, t2, IC, 1, 0);
  gemm(xubf, wq[8], bz3, t3, IC, 1, 0);
  hipLaunchKernelGGL(bilin_update_bf16, dim3(Bsz), blk, 0, stream, z, t1, t2, t3, zubf);

  // -------- 3-step LSTM (zero init, no bias) --------
  hipLaunchKernelGGL(zero_u32, dim3((unsigned)((U / 2 + 255) / 256)), blk, 0, stream,
                     (unsigned*)hbf, U / 2);                 // h = 0 (bf16)
  hipLaunchKernelGGL(zero_u32, dim3((unsigned)((U + 255) / 256)), blk, 0, stream,
                     (unsigned*)c, U);                       // c = 0 (f32)

  const unsigned short* seq[3]  = {xubf, yubf, zubf};
  float*                fout[3] = {t1, t2, t3};   // t's free after bilin stages
  for (int t = 0; t < 3; ++t) {
    gemm(seq[t], wihbf, nullptr, g, 4 * IC, 0, 0);
    gemm(hbf,    whhbf, nullptr, g, 4 * IC, 0, 1);   // g += h @ w_hh^T
    hipLaunchKernelGGL(lstm_gates, dim3(Bsz * IC / 256), blk, 0, stream,
                       g, hbf, c, fout[t]);
  }

  // -------- fused classifier heads + 4-way softmax --------
  hipLaunchKernelGGL(head_softmax, dim3((Bsz * 26 + 255) / 256), blk, 0, stream,
                     t1, t2, t3, wcx, bcx, wcy, bcy, wcz, bcz, wcxyz, bcxyz,
                     out, Bsz);
}
